// VQVAE_34892314313117
// MI455X (gfx1250) — matrix operand order repaired and checked by reading, not compiled, not run on hardware
//
#include <hip/hip_runtime.h>
#include <hip/hip_bf16.h>
#include <math.h>

typedef __attribute__((ext_vector_type(16))) _Float16 v16h;
typedef __attribute__((ext_vector_type(8)))  _Float16 v8h;
typedef __attribute__((ext_vector_type(2)))  _Float16 v2h;
typedef __attribute__((ext_vector_type(8)))  float    v8f;

// ---------------------------------------------------------------------------
// WMMA 16-bit A layout: lane half h, element e -> K = h*8+e (e<8),
// 16+h*8+(e-8) (e>=8): two contiguous 16B chunks. Decomposing K = ci*16+kh*4+kw
// over a 32-wide K tile kt: ci = 2*kt + c (c = chunk), kh = 2*h + s (s = sub),
// kw = 0..3 -> each 4-element group is 4 contiguous input pixels.
// B layout: lane col = lane%16, element e -> same K as A (two 16B chunks).
// ---------------------------------------------------------------------------
__device__ __forceinline__ v16h join8(v8h a0, v8h a1) {
  return __builtin_shufflevector(a0, a1, 0, 1, 2, 3, 4, 5, 6, 7,
                                 8, 9, 10, 11, 12, 13, 14, 15);
}

// ---------------------------------------------------------------------------
// Utility kernels
// ---------------------------------------------------------------------------
__global__ void zero_buf(float* __restrict__ p, int n) {
  int i = blockIdx.x * blockDim.x + threadIdx.x;
  if (i < n) p[i] = 0.0f;
}

// Zero the 1-pixel halo border of a padded f16 activation buffer [BC][Hp][Wp].
__global__ void zero_halo(_Float16* __restrict__ buf, int BC, int Hp, int Wp) {
  long i = (long)blockIdx.x * blockDim.x + threadIdx.x;
  long tot = (long)BC * Hp * Wp;
  if (i >= tot) return;
  int xy = (int)(i % (Hp * Wp));
  int y = xy / Wp, x = xy % Wp;
  if (y == 0 || y == Hp - 1 || x == 0 || x == Wp - 1) buf[i] = (_Float16)0.0f;
}

// Cast f32 NCHW -> f16 padded NCHW interior ([bc][y+1][x+1]).
__global__ void cast_pad_f32_f16(const float* __restrict__ in, _Float16* __restrict__ out,
                                 int H, int W, long n) {
  long i = (long)blockIdx.x * blockDim.x + threadIdx.x;
  if (i >= n) return;
  int hw = H * W;
  long bc = i / hw;
  int  xy = (int)(i % hw);
  int y = xy / W, x = xy % W;
  out[(bc * (H + 2) + y + 1) * (W + 2) + x + 1] = (_Float16)in[i];
}

__global__ void cast_f32_f16(const float* __restrict__ in, _Float16* __restrict__ out, long n) {
  long i = (long)blockIdx.x * blockDim.x + threadIdx.x;
  if (i < n) out[i] = (_Float16)in[i];
}

// Pack conv weight OIHW (Cout,Cin,4,4) -> f16 transposed BwT[n][k], k = ci*16+kh*4+kw.
__global__ void pack_w_conv(const float* __restrict__ w, _Float16* __restrict__ BwT,
                            int Cin, int Cout, int K, int Kpad, int Npad) {
  int i = blockIdx.x * blockDim.x + threadIdx.x;
  int tot = Kpad * Npad;
  if (i >= tot) return;
  int n = i / Kpad, k = i % Kpad;
  float v = 0.0f;
  if (k < K && n < Cout) {
    int ci = k >> 4, kh = (k >> 2) & 3, kw = k & 3;
    v = w[((n * Cin + ci) * 4 + kh) * 4 + kw];
  }
  BwT[i] = (_Float16)v;
}

// Pack deconv weight (Cin,Cout,4,4): flipped + swapped -> BwT[n][k] = w[ci][n][3-kh][3-kw].
__global__ void pack_w_deconv(const float* __restrict__ w, _Float16* __restrict__ BwT,
                              int Cin, int Cout, int K, int Kpad, int Npad) {
  int i = blockIdx.x * blockDim.x + threadIdx.x;
  int tot = Kpad * Npad;
  if (i >= tot) return;
  int n = i / Kpad, k = i % Kpad;
  float v = 0.0f;
  if (k < K && n < Cout) {
    int ci = k >> 4, kh = (k >> 2) & 3, kw = k & 3;
    v = w[((ci * Cout + n) * 4 + (3 - kh)) * 4 + (3 - kw)];
  }
  BwT[i] = (_Float16)v;
}

// ---------------------------------------------------------------------------
// Implicit-GEMM conv / deconv via v_wmma_f32_16x16x32_f16.
// Input is halo-padded [Cin][Hin+2][Win+2] f16 -> no bounds checks.
// Each wave owns a 16(M) x 32(N) C tile; B fragments are 16B-aligned direct
// global loads from transposed packed weights (hot in WGP$/L2); no barriers.
// ---------------------------------------------------------------------------
template <bool DECONV>
__global__ __launch_bounds__(256) void conv_gemm_wmma(
    const _Float16* __restrict__ act, const _Float16* __restrict__ BwT,
    const float* __restrict__ bias,
    _Float16* __restrict__ out16, float* __restrict__ out32,
    int Cin, int Hin, int Win, int Cout, int Ho, int Wo,
    int Kpad, int opad, int actmode)
{
  const int lane = threadIdx.x & 31;
  const int wave = threadIdx.x >> 5;
  const int half = lane >> 4;
  const int lm   = lane & 15;
  const int HoWo = Ho * Wo;
  const int Win2 = Win + 2;
  const long chstr = (long)(Hin + 2) * Win2;       // padded channel stride
  const int nbase = blockIdx.y * 32;
  const long m_tile = (long)blockIdx.x * 8 + wave;

  // Per-lane A row
  const long mA = m_tile * 16 + lm;
  const int bA  = (int)(mA / HoWo);
  const int pA  = (int)(mA % HoWo);
  const int oyA = pA / Wo;
  const int oxA = pA % Wo;

  // Per-(c,s) group base pointers at kt = 0 (advance by 2*chstr per K step).
  const _Float16* gp[2][2];
  bool rowok[2];
  int kw0 = 0;
#pragma unroll
  for (int s = 0; s < 2; ++s) {
    int kh = 2 * half + s;
    long row, col;
    if (!DECONV) {
      row = 2 * oyA + kh;          // padded row index, always in bounds
      col = 2 * oxA;               // 4 contiguous taps
      rowok[s] = true;
    } else {
      rowok[s] = (((oyA + kh) & 1) == 0);
      row = ((oyA + kh - 2) >> 1) + 1;   // safe even when !rowok
      kw0 = oxA & 1;
      col = ((oxA + kw0 - 2) >> 1) + 1;  // two taps: col, col+1
    }
#pragma unroll
    for (int c = 0; c < 2; ++c)
      gp[c][s] = act + ((long)bA * Cin + c) * chstr + row * Win2 + col;
  }

  // B fragment row pointers (two 16B chunks at half*8 and half*8+16).
  const _Float16* brow0 = &BwT[(long)(nbase + lm) * Kpad + half * 8];
  const _Float16* brow1 = &BwT[(long)(nbase + 16 + lm) * Kpad + half * 8];

  v8f acc0 = {}, acc1 = {};
  const int ktiles = Kpad >> 5;
  for (int kt = 0; kt < ktiles; ++kt) {
    const long kadv = (long)kt * 2 * chstr;

    v16h a;
#pragma unroll
    for (int c = 0; c < 2; ++c) {
#pragma unroll
      for (int s = 0; s < 2; ++s) {
        const int eb = c * 8 + s * 4;
        const _Float16* pp = gp[c][s] + kadv;
        if (!DECONV) {
          // 4 contiguous halves, 4B aligned: two b32 loads
          unsigned u0 = *(const unsigned*)pp;
          unsigned u1 = *(const unsigned*)(pp + 2);
          v2h p0 = __builtin_bit_cast(v2h, u0);
          v2h p1 = __builtin_bit_cast(v2h, u1);
          a[eb + 0] = p0.x; a[eb + 1] = p0.y;
          a[eb + 2] = p1.x; a[eb + 3] = p1.y;
        } else {
          _Float16 z = (_Float16)0.0f;
          _Float16 v0 = pp[0], v1 = pp[1];     // unconditional, always in bounds
          _Float16 lo = rowok[s] ? v0 : z;
          _Float16 hi = rowok[s] ? v1 : z;
          bool kodd = (kw0 != 0);
          a[eb + 0] = kodd ? z : lo;
          a[eb + 1] = kodd ? lo : z;
          a[eb + 2] = kodd ? z : hi;
          a[eb + 3] = kodd ? hi : z;
        }
      }
    }

    v16h b0 = join8(*(const v8h*)&brow0[kt * 32], *(const v8h*)&brow0[kt * 32 + 16]);
    v16h b1 = join8(*(const v8h*)&brow1[kt * 32], *(const v8h*)&brow1[kt * 32 + 16]);

    acc0 = __builtin_amdgcn_wmma_f32_16x16x32_f16(false, a, false, b0,
                                                  (short)0, acc0, false, false);
    acc1 = __builtin_amdgcn_wmma_f32_16x16x32_f16(false, a, false, b1,
                                                  (short)0, acc1, false, false);
  }

  // Epilogue: vgpr r holds M = half*8 + r; cols nbase+lm, nbase+16+lm.
  const int nc0 = nbase + lm;
  const int nc1 = nbase + 16 + lm;
  const float bi0 = (nc0 < Cout) ? bias[nc0] : 0.0f;
  const float bi1 = (nc1 < Cout) ? bias[nc1] : 0.0f;
  const int Hp = Ho + 2 * opad, Wp = Wo + 2 * opad;
  long mrow0 = m_tile * 16 + half * 8;
  int b2 = (int)(mrow0 / HoWo);
  int p2 = (int)(mrow0 % HoWo);
  int oy2 = p2 / Wo, ox2 = p2 % Wo;
#pragma unroll
  for (int r = 0; r < 8; ++r) {
    long base0 = (((long)b2 * Cout + nc0) * Hp + oy2 + opad) * Wp + ox2 + opad;
    long base1 = (((long)b2 * Cout + nc1) * Hp + oy2 + opad) * Wp + ox2 + opad;
    float v0 = acc0[r] + bi0;
    float v1 = acc1[r] + bi1;
    if (actmode == 0) {
      v0 = fmaxf(v0, 0.0f); v1 = fmaxf(v1, 0.0f);
    } else if (actmode == 2) {
      v0 = 1.0f / (1.0f + __expf(-v0)); v1 = 1.0f / (1.0f + __expf(-v1));
    }
    if (out16) {
      if (nc0 < Cout) out16[base0] = (_Float16)v0;
      if (nc1 < Cout) out16[base1] = (_Float16)v1;
    } else {
      if (nc0 < Cout) out32[base0] = v0;
      if (nc1 < Cout) out32[base1] = v1;
    }
    if (++ox2 == Wo) { ox2 = 0; if (++oy2 == Ho) { oy2 = 0; ++b2; } }
  }
}

// ---------------------------------------------------------------------------
// VQ distance GEMM: dist[b][j] = ||z_b||^2 - 2 (Z E^T)[b][j] + ||e_j||^2.
// All 4 M-tiles (M=64) in per-wave accumulators -> each emb element read once.
// Z staged in 256-wide K panels (32KB LDS): one barrier pair + one load/store
// clause covers 8 K tiles (32 WMMAs), amortizing barriers and stage waits.
// ---------------------------------------------------------------------------
__global__ __launch_bounds__(256) void vq_dist_wmma(
    const _Float16* __restrict__ z16, const _Float16* __restrict__ emb16,
    const float* __restrict__ znorm, const float* __restrict__ enorm,
    float* __restrict__ dist, int Kdim, int Ncode)
{
  __shared__ __align__(16) _Float16 Az[64 * 256];   // [m][k-panel], 32KB
  const int lane = threadIdx.x & 31;
  const int wave = threadIdx.x >> 5;
  const int half = lane >> 4;
  const int lm   = lane & 15;
  const int n_tile = blockIdx.x * 8 + wave;
  const int ncol = n_tile * 16 + lm;
  const bool nvalid = (ncol < Ncode);
  const _Float16* erow = &emb16[(long)(nvalid ? ncol : 0) * Kdim + half * 16];

  const int sm = threadIdx.x >> 2;         // 0..63 (row)
  const int sk = (threadIdx.x & 3) * 8;    // 0,8,16,24
  const _Float16* zrow = &z16[(long)sm * Kdim + sk];

  v8f acc[4] = {{}, {}, {}, {}};
  const int panels = Kdim >> 8;            // 12544 / 256 = 49
  for (int pt = 0; pt < panels; ++pt) {
    __syncthreads();
    // Stage 64 x 256 halves: 8x (global_load_b128 -> ds_store_b128) per thread
    v8h tmp[8];
#pragma unroll
    for (int j = 0; j < 8; ++j) tmp[j] = *(const v8h*)&zrow[pt * 256 + j * 32];
#pragma unroll
    for (int j = 0; j < 8; ++j) *(v8h*)&Az[sm * 256 + j * 32 + sk] = tmp[j];
    __syncthreads();

    if (pt + 1 < panels)
      __builtin_prefetch(&erow[(pt + 1) * 256], 0, 1);

#pragma unroll
    for (int kk = 0; kk < 8; ++kk) {
      v16h b = *(const v16h*)&erow[pt * 256 + kk * 32];   // 2x global_load_b128
#pragma unroll
      for (int mt = 0; mt < 4; ++mt) {
        const int ro = (mt * 16 + lm) * 256 + kk * 32 + half * 16;
        v8h a0 = *(const v8h*)&Az[ro];
        v8h a1 = *(const v8h*)&Az[ro + 8];
        v16h a = join8(a0, a1);
        acc[mt] = __builtin_amdgcn_wmma_f32_16x16x32_f16(false, a, false, b,
                                                         (short)0, acc[mt], false, false);
      }
    }
  }

  if (nvalid) {
#pragma unroll
    for (int mt = 0; mt < 4; ++mt) {
#pragma unroll
      for (int r = 0; r < 8; ++r) {
        int row = mt * 16 + half * 8 + r;
        dist[(long)row * Ncode + ncol] = znorm[row] - 2.0f * acc[mt][r] + enorm[ncol];
      }
    }
  }
}

// ---------------------------------------------------------------------------
// Reductions / VQ helpers
// ---------------------------------------------------------------------------
__global__ __launch_bounds__(256) void rownorm_f32(const float* __restrict__ src,
                                                   float* __restrict__ out, int Kdim) {
  __shared__ float sd[256];
  long row = blockIdx.x;
  float s = 0.0f;
  for (int j = threadIdx.x; j < Kdim; j += 256) {
    float v = src[row * Kdim + j];
    s += v * v;
  }
  sd[threadIdx.x] = s;
  __syncthreads();
  for (int st = 128; st > 0; st >>= 1) {
    if (threadIdx.x < st) sd[threadIdx.x] += sd[threadIdx.x + st];
    __syncthreads();
  }
  if (threadIdx.x == 0) out[row] = sd[0];
}

__global__ __launch_bounds__(256) void rownorm_f16(const _Float16* __restrict__ src,
                                                   float* __restrict__ out, int Kdim) {
  __shared__ float sd[256];
  long row = blockIdx.x;
  float s = 0.0f;
  for (int j = threadIdx.x; j < Kdim; j += 256) {
    float v = (float)src[row * Kdim + j];
    s += v * v;
  }
  sd[threadIdx.x] = s;
  __syncthreads();
  for (int st = 128; st > 0; st >>= 1) {
    if (threadIdx.x < st) sd[threadIdx.x] += sd[threadIdx.x + st];
    __syncthreads();
  }
  if (threadIdx.x == 0) out[row] = sd[0];
}

__global__ __launch_bounds__(256) void argmin_rows(const float* __restrict__ dist,
                                                   int* __restrict__ idx, int Ncode) {
  __shared__ float smin[256];
  __shared__ int   simn[256];
  int row = blockIdx.x;
  float best = 3.4e38f;
  int bi = 0;
  for (int j = threadIdx.x; j < Ncode; j += 256) {
    float v = dist[(long)row * Ncode + j];
    if (v < best) { best = v; bi = j; }
  }
  smin[threadIdx.x] = best;
  simn[threadIdx.x] = bi;
  __syncthreads();
  for (int s = 128; s > 0; s >>= 1) {
    if (threadIdx.x < s) {
      float o = smin[threadIdx.x + s];
      if (o < smin[threadIdx.x] ||
          (o == smin[threadIdx.x] && simn[threadIdx.x + s] < simn[threadIdx.x])) {
        smin[threadIdx.x] = o;
        simn[threadIdx.x] = simn[threadIdx.x + s];
      }
    }
    __syncthreads();
  }
  if (threadIdx.x == 0) idx[row] = simn[0];
}

// Gather quantized rows: dense q16d (for loss) + halo-padded q16p (for decoder).
__global__ void gather_q(const float* __restrict__ emb, const int* __restrict__ idx,
                         _Float16* __restrict__ q16d, _Float16* __restrict__ q16p,
                         int Kdim) {
  long i = (long)blockIdx.x * blockDim.x + threadIdx.x;
  long n = 64L * Kdim;
  if (i >= n) return;
  int b = (int)(i / Kdim);
  int d = (int)(i % Kdim);
  _Float16 v = (_Float16)emb[(long)idx[b] * Kdim + d];
  q16d[i] = v;
  int c = d / 196, xy = d % 196;
  int y = xy / 14, x = xy % 14;
  q16p[(((long)b * 64 + c) * 16 + y + 1) * 16 + x + 1] = v;
}

__global__ __launch_bounds__(256) void sqdiff_reduce_f16(const _Float16* __restrict__ a,
                                                         const _Float16* __restrict__ b,
                                                         long n, float* __restrict__ accum) {
  __shared__ float sd[256];
  float s = 0.0f;
  for (long i = (long)blockIdx.x * blockDim.x + threadIdx.x; i < n;
       i += (long)gridDim.x * blockDim.x) {
    float d = (float)a[i] - (float)b[i];
    s += d * d;
  }
  sd[threadIdx.x] = s;
  __syncthreads();
  for (int st = 128; st > 0; st >>= 1) {
    if (threadIdx.x < st) sd[threadIdx.x] += sd[threadIdx.x + st];
    __syncthreads();
  }
  if (threadIdx.x == 0) atomicAdd(accum, sd[0]);
}

__global__ __launch_bounds__(256) void sqdiff_reduce_f32(const float* __restrict__ a,
                                                         const float* __restrict__ b,
                                                         long n, float* __restrict__ accum) {
  __shared__ float sd[256];
  float s = 0.0f;
  for (long i = (long)blockIdx.x * blockDim.x + threadIdx.x; i < n;
       i += (long)gridDim.x * blockDim.x) {
    float d = a[i] - b[i];
    s += d * d;
  }
  sd[threadIdx.x] = s;
  __syncthreads();
  for (int st = 128; st > 0; st >>= 1) {
    if (threadIdx.x < st) sd[threadIdx.x] += sd[threadIdx.x + st];
    __syncthreads();
  }
  if (threadIdx.x == 0) atomicAdd(accum, sd[0]);
}

__global__ void finalize_losses(const float* __restrict__ scal, float* __restrict__ tail) {
  float mse_qz = scal[0] / (64.0f * 12544.0f);
  float vq = 1.25f * mse_qz;                // q_latent + 0.25*e_latent, numerically equal
  float recon = scal[1] / 9633792.0f;       // 64*3*224*224
  tail[0] = recon + vq;                     // loss
  tail[1] = recon;                          // recon_loss
  tail[2] = vq;                             // vq_loss
}

// ---------------------------------------------------------------------------
// Host launcher
// ---------------------------------------------------------------------------
extern "C" void kernel_launch(void* const* d_in, const int* in_sizes, int n_in,
                              void* d_out, int out_size, void* d_ws, size_t ws_size,
                              hipStream_t stream) {
  (void)in_sizes; (void)n_in; (void)out_size; (void)ws_size;

  const float* x   = (const float*)d_in[0];
  const float* ew1 = (const float*)d_in[1];  const float* eb1 = (const float*)d_in[2];
  const float* ew2 = (const float*)d_in[3];  const float* eb2 = (const float*)d_in[4];
  const float* ew3 = (const float*)d_in[5];  const float* eb3 = (const float*)d_in[6];
  const float* ew4 = (const float*)d_in[7];  const float* eb4 = (const float*)d_in[8];
  const float* emb = (const float*)d_in[9];
  const float* dw1 = (const float*)d_in[10]; const float* db1 = (const float*)d_in[11];
  const float* dw2 = (const float*)d_in[12]; const float* db2 = (const float*)d_in[13];
  const float* dw3 = (const float*)d_in[14]; const float* db3 = (const float*)d_in[15];
  const float* dw4 = (const float*)d_in[16]; const float* db4 = (const float*)d_in[17];

  char* p = (char*)d_ws;
  auto alloc = [&](size_t bytes) -> void* {
    void* r = (void*)p;
    p += (bytes + 255) & ~(size_t)255;
    return r;
  };
  _Float16* actA  = (_Float16*)alloc((size_t)56 << 20);
  _Float16* actB  = (_Float16*)alloc((size_t)56 << 20);
  _Float16* emb16 = (_Float16*)alloc((size_t)4000 * 12544 * 2);
  _Float16* z16   = (_Float16*)alloc((size_t)64 * 12544 * 2);
  _Float16* q16d  = (_Float16*)alloc((size_t)64 * 12544 * 2);
  _Float16* q16p  = (_Float16*)alloc((size_t)64 * 64 * 16 * 16 * 2);
  float*    dist  = (float*)alloc((size_t)64 * 4000 * 4);
  float*    enorm = (float*)alloc(4000 * 4);
  float*    znorm = (float*)alloc(64 * 4);
  int*      idx   = (int*)alloc(64 * 4);
  float*    scal  = (float*)alloc(256);
  _Float16* BwE1 = (_Float16*)alloc((size_t)64 * 32 * 2);
  _Float16* BwE2 = (_Float16*)alloc((size_t)512 * 64 * 2);
  _Float16* BwE3 = (_Float16*)alloc((size_t)1024 * 128 * 2);
  _Float16* BwE4 = (_Float16*)alloc((size_t)2048 * 64 * 2);
  _Float16* BwD1 = (_Float16*)alloc((size_t)1024 * 128 * 2);
  _Float16* BwD2 = (_Float16*)alloc((size_t)2048 * 64 * 2);
  _Float16* BwD3 = (_Float16*)alloc((size_t)1024 * 32 * 2);
  _Float16* BwD4 = (_Float16*)alloc((size_t)512 * 32 * 2);

  const long nx = 64L * 3 * 224 * 224;      // 9,633,792
  const long ne = 4000L * 12544;            // 50,176,000
  const long nq = 64L * 12544;              // 802,816

  auto gsz = [](long n) { return (unsigned)((n + 255) / 256); };
  auto halo = [&](_Float16* buf, int BC, int Hp, int Wp) {
    zero_halo<<<gsz((long)BC * Hp * Wp), 256, 0, stream>>>(buf, BC, Hp, Wp);
  };

  zero_buf<<<1, 64, 0, stream>>>(scal, 8);

  // x -> padded f16 in actA
  halo(actA, 64 * 3, 226, 226);
  cast_pad_f32_f16<<<gsz(nx), 256, 0, stream>>>(x, actA, 224, 224, nx);
  cast_f32_f16<<<gsz(ne), 256, 0, stream>>>(emb, emb16, ne);

  pack_w_conv  <<<gsz(64 * 32),    256, 0, stream>>>(ew1, BwE1, 3,   32, 48,   64,   32);
  pack_w_conv  <<<gsz(512 * 64),   256, 0, stream>>>(ew2, BwE2, 32,  64, 512,  512,  64);
  pack_w_conv  <<<gsz(1024 * 128), 256, 0, stream>>>(ew3, BwE3, 64, 128, 1024, 1024, 128);
  pack_w_conv  <<<gsz(2048 * 64),  256, 0, stream>>>(ew4, BwE4, 128, 64, 2048, 2048, 64);
  pack_w_deconv<<<gsz(1024 * 128), 256, 0, stream>>>(dw1, BwD1, 64, 128, 1024, 1024, 128);
  pack_w_deconv<<<gsz(2048 * 64),  256, 0, stream>>>(dw2, BwD2, 128, 64, 2048, 2048, 64);
  pack_w_deconv<<<gsz(1024 * 32),  256, 0, stream>>>(dw3, BwD3, 64,  32, 1024, 1024, 32);
  pack_w_deconv<<<gsz(512 * 32),   256, 0, stream>>>(dw4, BwD4, 32,   3, 512,  512,  32);

  rownorm_f32<<<4000, 256, 0, stream>>>(emb, enorm, 12544);

  auto launch = [&](bool deconv, const _Float16* act, const _Float16* BwT,
                    const float* bias, _Float16* o16, float* o32,
                    int Cin, int Hin, int Cout, int Ho, int K, int Npad,
                    int opad, int actmode) {
    int Kpad = (K + 31) & ~31;
    long M = 64L * Ho * Ho;
    dim3 grid((unsigned)(M / 128), (unsigned)(Npad / 32));
    if (deconv)
      conv_gemm_wmma<true><<<grid, 256, 0, stream>>>(act, BwT, bias, o16, o32,
          Cin, Hin, Hin, Cout, Ho, Ho, Kpad, opad, actmode);
    else
      conv_gemm_wmma<false><<<grid, 256, 0, stream>>>(act, BwT, bias, o16, o32,
          Cin, Hin, Hin, Cout, Ho, Ho, Kpad, opad, actmode);
  };

  // Encoder
  halo(actB, 64 * 32, 114, 114);
  launch(false, actA, BwE1, eb1, actB, nullptr, 3,   224, 32, 112, 48,   32,  1, 0);
  halo(actA, 64 * 64, 58, 58);
  launch(false, actB, BwE2, eb2, actA, nullptr, 32,  112, 64, 56,  512,  64,  1, 0);
  halo(actB, 64 * 128, 30, 30);
  launch(false, actA, BwE3, eb3, actB, nullptr, 64,  56, 128, 28,  1024, 128, 1, 0);
  launch(false, actB, BwE4, eb4, z16,  nullptr, 128, 28,  64, 14,  2048, 64,  0, 1);

  // VQ
  rownorm_f16<<<64, 256, 0, stream>>>(z16, znorm, 12544);
  vq_dist_wmma<<<32, 256, 0, stream>>>(z16, emb16, znorm, enorm, dist, 12544, 4000);
  argmin_rows<<<64, 256, 0, stream>>>(dist, idx, 4000);
  halo(q16p, 64 * 64, 16, 16);
  gather_q<<<gsz(nq), 256, 0, stream>>>(emb, idx, q16d, q16p, 12544);
  sqdiff_reduce_f16<<<512, 256, 0, stream>>>(q16d, z16, nq, scal + 0);

  // Decoder
  float* xr = (float*)d_out;
  halo(actA, 64 * 128, 30, 30);
  launch(true, q16p, BwD1, db1, actA, nullptr, 64,  14, 128, 28,  1024, 128, 1, 0);
  halo(actB, 64 * 64, 58, 58);
  launch(true, actA, BwD2, db2, actB, nullptr, 128, 28,  64, 56,  2048, 64,  1, 0);
  halo(actA, 64 * 32, 114, 114);
  launch(true, actB, BwD3, db3, actA, nullptr, 64,  56,  32, 112, 1024, 32,  1, 0);
  launch(true, actA, BwD4, db4, nullptr, xr,   32, 112,  3,  224, 512,  32,  0, 2);

  // Losses
  sqdiff_reduce_f32<<<2048, 256, 0, stream>>>(xr, x, nx, scal + 1);
  finalize_losses<<<1, 1, 0, stream>>>(scal, xr + nx);
}
